// GQA_kernel_67207648247889
// MI455X (gfx1250) — matrix operand order repaired and checked by reading, not compiled, not run on hardware
//
#include <hip/hip_runtime.h>

typedef __attribute__((ext_vector_type(16))) _Float16 v16h;
typedef __attribute__((ext_vector_type(8)))  _Float16 v8h;
typedef __attribute__((ext_vector_type(8)))  float    v8f;

union AF { v16h v; v8h h[2]; _Float16 e[16]; };
union F8 { v8f v; float f[8]; };
union H8 { v8h h; _Float16 e[8]; };
union P2 { unsigned u; _Float16 e[2]; };

constexpr int S_LEN = 2048;   // sequence length
constexpr int NH    = 32;     // query heads
constexpr int NKV   = 8;      // kv heads
constexpr int HD    = 128;    // head dim
constexpr int BM    = 128;    // query rows per block (8 waves x 16)
constexpr int BN    = 32;     // key cols per iteration
constexpr int NW    = 8;      // waves per block
// 1/sqrt(128) * log2(e): softmax in exp2 domain
constexpr float QK_SCALE = 0.08838834764831845f * 1.4426950408889634f;

// gfx1250 async copy: global -> LDS, 16B per lane, tracked with ASYNCcnt
__device__ __forceinline__ void async_b128_to_lds(const _Float16* gp, _Float16* lp) {
  unsigned lds = (unsigned)(size_t)lp;                 // low 32 bits = LDS offset
  unsigned long long ga = (unsigned long long)(size_t)gp;
  asm volatile("global_load_async_to_lds_b128 %0, %1, off"
               :: "v"(lds), "v"(ga) : "memory");
}
__device__ __forceinline__ void wait_async0() {
  asm volatile("s_wait_asynccnt 0x0" ::: "memory");
}

// dpp8 selector for lane ^= m within 8-lane groups
constexpr int dpp8_xor(int m) {
  int s = 0;
  for (int i = 0; i < 8; ++i) s |= ((i ^ m) & 7) << (3 * i);
  return s;
}
template <int SEL>
__device__ __forceinline__ float fmax_xor_dpp8(float x) {
  int xi = __builtin_bit_cast(int, x);
  int yi = __builtin_amdgcn_mov_dpp8(xi, SEL);
  return fmaxf(x, __builtin_bit_cast(float, yi));
}

__global__ __launch_bounds__(256)
void gqa_flash_fwd(const _Float16* __restrict__ Q,
                   const _Float16* __restrict__ K,
                   const _Float16* __restrict__ V,
                   _Float16*       __restrict__ O)
{
  // double-buffered K/V tiles; rows padded 8 halfs (16B) for b128 alignment
  __shared__ _Float16 sK [2][BN][HD + 8];    // K tile, row major [key][d]
  __shared__ _Float16 sVt[2][HD][BN + 8];    // V tile, transposed [d][key]
  __shared__ _Float16 sP [NW][16][BN + 8];   // per-wave P relayout scratch

  const int tid   = threadIdx.x;
  const int wave  = tid >> 5;
  const int lane  = tid & 31;
  const int lo16  = lane & 15;
  const int hi    = lane >> 4;
  const int k0sel = hi << 3;
  const int h     = blockIdx.y;
  const int hk    = h >> 2;                    // kv head = h / 4
  const int q0    = blockIdx.x * BM;
  const int qrow  = q0 + wave * 16 + lo16;     // A-frag row
  const int rowg  = q0 + wave * 16 + (hi << 3);// C-frag row base
  const int rl    = hi << 3;
  // wave-uniform causal bound, made scalar so the skip is an SCC branch
  const int wave_last_row = __builtin_amdgcn_readfirstlane(q0 + wave * 16 + 15);

  // ---- resident Q fragments (A layout), 4 chunks along Dqk; streamed (NT) ----
  AF qf[4];
  {
    const _Float16* qp = Q + ((size_t)qrow * NH + h) * HD;
    #pragma unroll
    for (int c = 0; c < 4; ++c) {
      qf[c].h[0] = __builtin_nontemporal_load((const v8h*)(qp + c * 32 + (hi << 4)));
      qf[c].h[1] = __builtin_nontemporal_load((const v8h*)(qp + c * 32 + (hi << 4) + 8));
    }
  }

  // all-ones B fragment: row-sum of P via WMMA (P . 1), lands in C layout
  AF onesf;
  #pragma unroll
  for (int e = 0; e < 16; ++e) onesf.e[e] = (_Float16)1.0f;

  F8 acc[8];
  #pragma unroll
  for (int f = 0; f < 8; ++f)
    #pragma unroll
    for (int i = 0; i < 8; ++i) acc[f].f[i] = 0.0f;

  float mrow[8], lrow[8];
  #pragma unroll
  for (int i = 0; i < 8; ++i) { mrow[i] = -1e30f; lrow[i] = 0.0f; }

  // staging indices: K = 32 rows x 8 chunks of 16 halfs; V = 16 row-pairs x 16 d-chunks
  const int srow   = tid >> 3;
  const int schunk = (tid & 7) * 16;
  const int vkr    = (tid & 15) * 2;     // key row pair
  const int vd0    = (tid >> 4) * 8;     // d chunk

  const int ntile = (q0 + BM) / BN;

  // ---- prologue: stage tile 0 into buffer 0 ----
  {
    const _Float16* kp = K + ((size_t)srow * NKV + hk) * HD + schunk;
    async_b128_to_lds(kp,     &sK[0][srow][schunk]);
    async_b128_to_lds(kp + 8, &sK[0][srow][schunk + 8]);
    H8 a, b;
    a.h = *(const v8h*)(V + ((size_t)vkr * NKV + hk) * HD + vd0);
    b.h = *(const v8h*)(V + ((size_t)(vkr + 1) * NKV + hk) * HD + vd0);
    #pragma unroll
    for (int j = 0; j < 8; ++j) {
      P2 p; p.e[0] = a.e[j]; p.e[1] = b.e[j];
      *(unsigned*)&sVt[0][vd0 + j][vkr] = p.u;
    }
  }
  wait_async0();
  __syncthreads();

  for (int it = 0; it < ntile; ++it) {
    const int  buf  = it & 1;
    const int  kt0  = it * BN;
    const bool more = (it + 1 < ntile);

    // ---- issue next tile's loads (async K -> LDS, V -> regs) ----
    H8 vr0, vr1;
    if (more) {
      const int kn = kt0 + BN;
      const _Float16* kp = K + ((size_t)(kn + srow) * NKV + hk) * HD + schunk;
      async_b128_to_lds(kp,     &sK[buf ^ 1][srow][schunk]);
      async_b128_to_lds(kp + 8, &sK[buf ^ 1][srow][schunk + 8]);
      vr0.h = *(const v8h*)(V + ((size_t)(kn + vkr) * NKV + hk) * HD + vd0);
      vr1.h = *(const v8h*)(V + ((size_t)(kn + vkr + 1) * NKV + hk) * HD + vd0);
    }

    if (kt0 <= wave_last_row) {   // scalar causal skip (EXEC stays all-ones)
      // ---- preload all K B-fragments, then two interleaved WMMA chains ----
      AF kb[2][4];
      #pragma unroll
      for (int nt = 0; nt < 2; ++nt)
        #pragma unroll
        for (int c = 0; c < 4; ++c) {
          const _Float16* p = &sK[buf][nt * 16 + lo16][c * 32 + (hi << 4)];
          kb[nt][c].h[0] = *(const v8h*)p;
          kb[nt][c].h[1] = *(const v8h*)(p + 8);
        }
      F8 sc[2];
      #pragma unroll
      for (int nt = 0; nt < 2; ++nt) {
        v8f a = {0.f, 0.f, 0.f, 0.f, 0.f, 0.f, 0.f, 0.f};
        #pragma unroll
        for (int c = 0; c < 4; ++c)
          a = __builtin_amdgcn_wmma_f32_16x16x32_f16(false, qf[c].v, false, kb[nt][c].v,
                                                     (short)0, a, false, false);
        sc[nt].v = a;
      }

      // ---- causal mask + scale (exp2 domain) ----
      #pragma unroll
      for (int nt = 0; nt < 2; ++nt)
        #pragma unroll
        for (int i = 0; i < 8; ++i) {
          float sv = sc[nt].f[i] * QK_SCALE;
          sc[nt].f[i] = ((kt0 + nt * 16 + lo16) > (rowg + i)) ? -1e30f : sv;
        }

      // ---- row max across the 16-lane half: 3x dpp8 (VALU) + 1x shfl (DS) ----
      float pm[8];
      #pragma unroll
      for (int i = 0; i < 8; ++i) {
        float t = fmaxf(sc[0].f[i], sc[1].f[i]);
        t = fmax_xor_dpp8<dpp8_xor(1)>(t);
        t = fmax_xor_dpp8<dpp8_xor(2)>(t);
        t = fmax_xor_dpp8<dpp8_xor(4)>(t);
        t = fmaxf(t, __shfl_xor(t, 8, 32));
        pm[i] = t;
      }

      // ---- online rescale; P (f16) to wave-private LDS ----
      float alpha[8];
      #pragma unroll
      for (int i = 0; i < 8; ++i) {
        float mnew = fmaxf(mrow[i], pm[i]);
        alpha[i]   = exp2f(mrow[i] - mnew);
        mrow[i]    = mnew;
        float p0 = exp2f(sc[0].f[i] - mnew);
        float p1 = exp2f(sc[1].f[i] - mnew);
        sP[wave][rl + i][lo16]      = (_Float16)p0;
        sP[wave][rl + i][lo16 + 16] = (_Float16)p1;
        #pragma unroll
        for (int f = 0; f < 8; ++f) acc[f].f[i] *= alpha[i];
      }

      // ---- reload P as A-fragment (same-wave DS ordering) ----
      AF pf;
      {
        const _Float16* pp = &sP[wave][lo16][0];
        pf.h[0] = *(const v8h*)(pp + (hi << 4));
        pf.h[1] = *(const v8h*)(pp + (hi << 4) + 8);
      }

      // ---- row sum of P via ones-matmul: exact C-layout match, no shuffles ----
      F8 lsum;
      {
        v8f z = {0.f, 0.f, 0.f, 0.f, 0.f, 0.f, 0.f, 0.f};
        lsum.v = __builtin_amdgcn_wmma_f32_16x16x32_f16(false, pf.v, false, onesf.v,
                                                        (short)0, z, false, false);
      }
      #pragma unroll
      for (int i = 0; i < 8; ++i) lrow[i] = lrow[i] * alpha[i] + lsum.f[i];

      // ---- O += P V : preload 4 B-frags per group, independent accumulators ----
      #pragma unroll
      for (int g = 0; g < 2; ++g) {
        AF vbf[4];
        #pragma unroll
        for (int f = 0; f < 4; ++f) {
          const _Float16* p = &sVt[buf][(g * 4 + f) * 16 + lo16][hi << 4];
          vbf[f].h[0] = *(const v8h*)p;
          vbf[f].h[1] = *(const v8h*)(p + 8);
        }
        #pragma unroll
        for (int f = 0; f < 4; ++f)
          acc[g * 4 + f].v = __builtin_amdgcn_wmma_f32_16x16x32_f16(
              false, pf.v, false, vbf[f].v, (short)0, acc[g * 4 + f].v, false, false);
      }
    }

    // ---- commit next V tile (transposed, b32 pair stores) into other buffer ----
    if (more) {
      #pragma unroll
      for (int j = 0; j < 8; ++j) {
        P2 p; p.e[0] = vr0.e[j]; p.e[1] = vr1.e[j];
        *(unsigned*)&sVt[buf ^ 1][vd0 + j][vkr] = p.u;
      }
    }
    wait_async0();
    __syncthreads();
  }

  // ---- normalize and store O (f16, NT streaming stores) ----
  #pragma unroll
  for (int i = 0; i < 8; ++i) {
    float inv = 1.0f / lrow[i];
    _Float16* op = O + ((size_t)(rowg + i) * NH + h) * HD + lo16;
    #pragma unroll
    for (int f = 0; f < 8; ++f)
      __builtin_nontemporal_store((_Float16)(acc[f].f[i] * inv), op + f * 16);
  }
}

extern "C" void kernel_launch(void* const* d_in, const int* in_sizes, int n_in,
                              void* d_out, int out_size, void* d_ws, size_t ws_size,
                              hipStream_t stream) {
  (void)in_sizes; (void)n_in; (void)out_size; (void)d_ws; (void)ws_size;
  const _Float16* Q = (const _Float16*)d_in[0];
  const _Float16* K = (const _Float16*)d_in[1];
  const _Float16* V = (const _Float16*)d_in[2];
  _Float16*       O = (_Float16*)d_out;
  dim3 grid(S_LEN / BM, NH);   // 16 x 32
  gqa_flash_fwd<<<grid, 256, 0, stream>>>(Q, K, V, O);
}
